// NPChangeTransitionPrior_39075612459509
// MI455X (gfx1250) — matrix operand order repaired and checked by reading, not compiled, not run on hardware
//
#include <hip/hip_runtime.h>
#include <math.h>

// ---------------------------------------------------------------------------
// Types / helpers
// ---------------------------------------------------------------------------
typedef __attribute__((ext_vector_type(16))) __bf16 v16bf;
typedef __attribute__((ext_vector_type(8)))  float  v8f;

static constexpr float SLOPE_ = 0.2f;
static constexpr int B_  = 64;
static constexpr int T_  = 512;
static constexpr int D_  = 8;
static constexpr int H_  = 128;
static constexpr int E_  = 96;
static constexpr int FG_ = 145;   // H + L*D + 1
static constexpr int TL_ = 510;   // T - L
static constexpr int N_  = B_ * TL_;          // 32640 = 255 * 128
static constexpr int TILES_ = N_ / 128;       // 255

__device__ __forceinline__ unsigned short f2bf(float f) {
  unsigned u = __float_as_uint(f);
  u += 0x7FFFu + ((u >> 16) & 1u);            // round-to-nearest-even
  return (unsigned short)(u >> 16);
}
__device__ __forceinline__ v8f vzero() {
  v8f z = {0.f, 0.f, 0.f, 0.f, 0.f, 0.f, 0.f, 0.f};
  return z;
}

// Vectorized f32 -> bf16 staging of a contiguous block (count % 4 == 0).
__device__ __forceinline__ void stage_cvt_vec4(const float* __restrict__ g,
                                               unsigned short* __restrict__ lds,
                                               int count, int tid) {
  for (int i = tid * 4; i < count; i += 256 * 4) {
    float4 f = *(const float4*)(g + i);
    uint2 p;
    p.x = (unsigned)f2bf(f.x) | ((unsigned)f2bf(f.y) << 16);
    p.y = (unsigned)f2bf(f.z) | ((unsigned)f2bf(f.w) << 16);
    *(uint2*)(lds + i) = p;
  }
}

union Frag32B { v16bf v; uint4 q[2]; };

// ---------------------------------------------------------------------------
// Wave-level GEMM micro-kernel: D(Rx16 cols) += A(RxK) * W^T, all bf16 in LDS.
// A: row-major, stride sA (bf16 elems).  W: [out][k] row-major, stride sW.
// A-frag (16-bit A 16x32): lane l: M=l&15; half=l>>4; chunk0 covers
// K = K0 + half*8 + [0..7], chunk1 covers K = K0 + 16 + half*8 + [0..7].
// B-frag (16-bit B 32x16): lane l: N=l&15; half=l>>4; chunk0 covers
// K = K0 + half*8 + [0..7], chunk1 covers K = K0 + 16 + half*8 + [0..7].
// C/D layout: lane l: N=l&15; vgpr v: M = v + 8*(l>>4).
// ---------------------------------------------------------------------------
template<int KSTEPS, int ROWTILES>
__device__ __forceinline__ void wmma_gemm(const unsigned short* __restrict__ A, int sA,
                                          const unsigned short* __restrict__ W, int sW,
                                          int lane, int colbase, v8f* acc) {
  const int lm   = lane & 15;
  const int half = lane >> 4;
#pragma unroll
  for (int ks = 0; ks < KSTEPS; ++ks) {
    const int K0 = ks * 32;
    Frag32B bfr;
    const unsigned short* wp = W + (colbase + lm) * sW + K0 + half * 8;
    bfr.q[0] = *(const uint4*)(wp);
    bfr.q[1] = *(const uint4*)(wp + 16);
#pragma unroll
    for (int r = 0; r < ROWTILES; ++r) {
      Frag32B afr;
      const unsigned short* ap = A + (r * 16 + lm) * sA + K0 + half * 8;
      afr.q[0] = *(const uint4*)(ap);
      afr.q[1] = *(const uint4*)(ap + 16);
      acc[r] = __builtin_amdgcn_wmma_f32_16x16x32_bf16(
          false, afr.v, false, bfr.v, (short)0, acc[r], false, false);
    }
  }
}

// ---------------------------------------------------------------------------
// Kernel 1: fc_mlp — emb_h = (lrelu(lrelu(e@W0^T+b0)@W1^T+b1))@Wf^T+bf
// One block = 128-row tile. 8 waves, wave w owns output cols [16w,16w+16).
// ---------------------------------------------------------------------------
__global__ __launch_bounds__(256) void fc_mlp_kernel(
    const float* __restrict__ emb,
    const float* __restrict__ W0, const float* __restrict__ b0,
    const float* __restrict__ W1, const float* __restrict__ b1,
    const float* __restrict__ Wf, const float* __restrict__ bfv,
    unsigned short* __restrict__ embh) {
  extern __shared__ char smem[];
  unsigned short* X  = (unsigned short*)smem;      // 128 x 96
  unsigned short* A1 = X  + 128 * 96;              // 128 x 128
  unsigned short* A2 = A1 + 128 * 128;             // 128 x 128
  unsigned short* Wl = A2 + 128 * 128;             // 128 x 128 (layer0 uses stride 96)
  float* bias = (float*)(Wl + 128 * 128);          // 128 f32

  const int tid  = threadIdx.x;
  const int lane = tid & 31;
  const int wave = tid >> 5;
  const int lm   = lane & 15;
  const int half = lane >> 4;
  const int n0   = blockIdx.x * 128;
  const int col  = wave * 16 + lm;

  __builtin_prefetch(W1 + tid * 16, 0, 0);
  __builtin_prefetch(Wf + tid * 16, 0, 0);

  stage_cvt_vec4(emb + (size_t)n0 * E_, X, 128 * E_, tid);  // rows contiguous
  stage_cvt_vec4(W0, Wl, H_ * E_, tid);
  if (tid < H_) bias[tid] = b0[tid];
  __syncthreads();

  v8f acc[8];
  // layer 0: K = 96
#pragma unroll
  for (int r = 0; r < 8; ++r) acc[r] = vzero();
  wmma_gemm<3, 8>(X, E_, Wl, E_, lane, wave * 16, acc);
#pragma unroll
  for (int r = 0; r < 8; ++r)
#pragma unroll
    for (int v = 0; v < 8; ++v) {
      int row = r * 16 + v + 8 * half;
      float z = acc[r][v] + bias[col];
      A1[row * H_ + col] = f2bf(z >= 0.f ? z : SLOPE_ * z);
    }
  __syncthreads();

  // layer 1: K = 128
  stage_cvt_vec4(W1, Wl, H_ * H_, tid);
  if (tid < H_) bias[tid] = b1[tid];
  __syncthreads();
#pragma unroll
  for (int r = 0; r < 8; ++r) acc[r] = vzero();
  wmma_gemm<4, 8>(A1, H_, Wl, H_, lane, wave * 16, acc);
#pragma unroll
  for (int r = 0; r < 8; ++r)
#pragma unroll
    for (int v = 0; v < 8; ++v) {
      int row = r * 16 + v + 8 * half;
      float z = acc[r][v] + bias[col];
      A2[row * H_ + col] = f2bf(z >= 0.f ? z : SLOPE_ * z);
    }
  __syncthreads();

  // final layer: linear, store bf16 emb_h to workspace
  stage_cvt_vec4(Wf, Wl, H_ * H_, tid);
  if (tid < H_) bias[tid] = bfv[tid];
  __syncthreads();
#pragma unroll
  for (int r = 0; r < 8; ++r) acc[r] = vzero();
  wmma_gemm<4, 8>(A2, H_, Wl, H_, lane, wave * 16, acc);
#pragma unroll
  for (int r = 0; r < 8; ++r)
#pragma unroll
    for (int v = 0; v < 8; ++v) {
      int row = r * 16 + v + 8 * half;
      embh[(size_t)(n0 + row) * H_ + col] = f2bf(acc[r][v] + bias[col]);
    }
}

// ---------------------------------------------------------------------------
// Kernel 2: g_mlp + JVP for one (128-row tile, d).
// LDS activation buffers hold 256 rows: rows [0,128) = h, rows [128,256) = dh,
// so the value GEMM and the tangent GEMM are one fused R=256 WMMA GEMM
// (the two chains share the same weight matrix). The final 128->1 layer is
// also WMMA: B has gWf in column 0, zeros elsewhere; 16 row-tiles split
// 2-per-wave so there is no serial scalar tail.
// ---------------------------------------------------------------------------
__global__ __launch_bounds__(256) void g_mlp_jvp_kernel(
    const float* __restrict__ x,
    const unsigned short* __restrict__ embh,
    const float* __restrict__ gW0, const float* __restrict__ gb0,
    const float* __restrict__ gW1, const float* __restrict__ gb1,
    const float* __restrict__ gW2, const float* __restrict__ gb2,
    const float* __restrict__ gWf, const float* __restrict__ gbf,
    float* __restrict__ resid, float* __restrict__ perlog) {
  extern __shared__ char smem[];
  unsigned short* X  = (unsigned short*)smem;      // 128 x 160  (Fg=145 zero-padded)
  unsigned short* A1 = X  + 128 * 160;             // 256 x 128  (h | dh)
  unsigned short* A2 = A1 + 256 * 128;             // 256 x 128
  unsigned short* Wl = A2 + 256 * 128;             // 128 x 160
  float* bias  = (float*)(Wl + 128 * 160);         // 128
  float* dpre0 = bias + 128;                       // 128 (layer-0 tangent preact)

  const int tid  = threadIdx.x;
  const int lane = tid & 31;
  const int wave = tid >> 5;
  const int lm   = lane & 15;
  const int half = lane >> 4;
  const int d    = blockIdx.y;
  const int n0   = blockIdx.x * 128;
  const int col  = wave * 16 + lm;

  __builtin_prefetch(gW1 + (size_t)d * H_ * H_ + tid * 16, 0, 0);
  __builtin_prefetch(gW2 + (size_t)d * H_ * H_ + tid * 16, 0, 0);

  // ---- Build input tile X = [emb_h(128) | yy(16) | xx_d(1) | pad(15)] ----
  // cols [0,128): bf16 copy from workspace, 8 elems (16B) per transfer
  for (int c = tid * 8; c < 128 * 128; c += 256 * 8) {
    int row = c >> 7, k = c & 127;
    *(uint4*)(X + row * 160 + k) =
        *(const uint4*)(embh + (size_t)(n0 + row) * H_ + k);
  }
  // cols [128,160): window data + zero pad
  for (int i = tid; i < 128 * 32; i += 256) {
    int row = i >> 5, c = 128 + (i & 31);
    int n = n0 + row;
    unsigned short v = 0;
    if (c < FG_) {
      int b = n / TL_, t = n - b * TL_;
      const float* xr = x + (size_t)(b * T_ + t) * D_;   // 17 usable floats
      v = f2bf(c < 144 ? xr[c - 128] : xr[16 + d]);
    }
    X[row * 160 + c] = v;
  }
  // Stage gW0[d] (128 x 145, zero-padded to 160), bias, layer-0 tangent preact
  for (int i = tid; i < 128 * 160; i += 256) {
    int row = i / 160, k = i % 160;
    Wl[i] = (k < FG_) ? f2bf(gW0[(size_t)(d * H_ + row) * FG_ + k]) : (unsigned short)0;
  }
  if (tid < H_) {
    bias[tid]  = gb0[d * H_ + tid];
    dpre0[tid] = gW0[(size_t)(d * H_ + tid) * FG_ + (FG_ - 1)];
  }
  __syncthreads();

  // ---- layer 0: R=128, K=160(145) ----
  v8f acc0[8];
#pragma unroll
  for (int r = 0; r < 8; ++r) acc0[r] = vzero();
  wmma_gemm<5, 8>(X, 160, Wl, 160, lane, wave * 16, acc0);
#pragma unroll
  for (int r = 0; r < 8; ++r)
#pragma unroll
    for (int v = 0; v < 8; ++v) {
      int row = r * 16 + v + 8 * half;
      float z = acc0[r][v] + bias[col];
      float m = z >= 0.f ? 1.f : SLOPE_;
      A1[row * H_ + col]          = f2bf(z >= 0.f ? z : SLOPE_ * z);   // h0
      A1[(128 + row) * H_ + col]  = f2bf(m * dpre0[col]);              // dh0
    }
  __syncthreads();

  // ---- layer 1: fused value+tangent GEMM, R=256, K=128 ----
  stage_cvt_vec4(gW1 + (size_t)d * H_ * H_, Wl, H_ * H_, tid);
  if (tid < H_) bias[tid] = gb1[d * H_ + tid];
  __syncthreads();
  v8f acc[16];
#pragma unroll
  for (int r = 0; r < 16; ++r) acc[r] = vzero();
  wmma_gemm<4, 16>(A1, H_, Wl, H_, lane, wave * 16, acc);
#pragma unroll
  for (int r = 0; r < 8; ++r)
#pragma unroll
    for (int v = 0; v < 8; ++v) {
      int row = r * 16 + v + 8 * half;
      float z = acc[r][v] + bias[col];
      float m = z >= 0.f ? 1.f : SLOPE_;
      A2[row * H_ + col]         = f2bf(z >= 0.f ? z : SLOPE_ * z);    // h1
      A2[(128 + row) * H_ + col] = f2bf(m * acc[r + 8][v]);            // dh1
    }
  __syncthreads();

  // ---- layer 2: fused value+tangent GEMM, R=256, K=128 ----
  stage_cvt_vec4(gW2 + (size_t)d * H_ * H_, Wl, H_ * H_, tid);
  if (tid < H_) bias[tid] = gb2[d * H_ + tid];
  __syncthreads();
#pragma unroll
  for (int r = 0; r < 16; ++r) acc[r] = vzero();
  wmma_gemm<4, 16>(A2, H_, Wl, H_, lane, wave * 16, acc);
#pragma unroll
  for (int r = 0; r < 8; ++r)
#pragma unroll
    for (int v = 0; v < 8; ++v) {
      int row = r * 16 + v + 8 * half;
      float z = acc[r][v] + bias[col];
      float m = z >= 0.f ? 1.f : SLOPE_;
      A1[row * H_ + col]         = f2bf(z >= 0.f ? z : SLOPE_ * z);    // h2
      A1[(128 + row) * H_ + col] = f2bf(m * acc[r + 8][v]);            // dh2
    }
  __syncthreads();

  // ---- final layer as WMMA: B col 0 = gWf, cols 1..15 = 0 ----
  for (int i = tid; i < 16 * H_; i += 256) {
    int row = i >> 7, k = i & 127;
    Wl[i] = (row == 0) ? f2bf(gWf[d * H_ + k]) : (unsigned short)0;
  }
  __syncthreads();
  v8f facc[2];
  facc[0] = vzero();
  facc[1] = vzero();
  // wave w handles row-tiles {2w, 2w+1} of the 256-row (h|dh) matrix, col-tile 0
  wmma_gemm<4, 2>(A1 + (size_t)wave * 32 * H_, H_, Wl, H_, lane, 0, facc);
  if (lm == 0) {
    float bf_d = gbf[d];
#pragma unroll
    for (int rt = 0; rt < 2; ++rt)
#pragma unroll
      for (int v = 0; v < 8; ++v) {
        int row = wave * 32 + rt * 16 + v + 8 * half;
        float val = facc[rt][v];
        if (row < 128)
          resid[(size_t)(n0 + row) * D_ + d] = val + bf_d;            // out
        else
          perlog[(size_t)(n0 + row - 128) * D_ + d] = logf(fabsf(val)); // log|pdd|
      }
  }
}

// ---------------------------------------------------------------------------
// Kernel 3: deterministic per-batch reduction of 510*8 log-abs-det terms.
// ---------------------------------------------------------------------------
__global__ __launch_bounds__(256) void logdet_reduce_kernel(
    const float* __restrict__ perlog, float* __restrict__ outlog) {
  __shared__ float sm[256];
  const int b = blockIdx.x;
  float s = 0.f;
  for (int i = threadIdx.x; i < TL_ * D_; i += 256)
    s += perlog[(size_t)b * TL_ * D_ + i];
  sm[threadIdx.x] = s;
  __syncthreads();
  for (int o = 128; o > 0; o >>= 1) {
    if (threadIdx.x < o) sm[threadIdx.x] += sm[threadIdx.x + o];
    __syncthreads();
  }
  if (threadIdx.x == 0) outlog[b] = sm[0];
}

// ---------------------------------------------------------------------------
// Host launcher
// ---------------------------------------------------------------------------
extern "C" void kernel_launch(void* const* d_in, const int* in_sizes, int n_in,
                              void* d_out, int out_size, void* d_ws, size_t ws_size,
                              hipStream_t stream) {
  (void)in_sizes; (void)n_in; (void)out_size; (void)ws_size;
  const float* x    = (const float*)d_in[0];
  const float* emb  = (const float*)d_in[1];
  const float* fcW0 = (const float*)d_in[2];
  const float* fcb0 = (const float*)d_in[3];
  const float* fcW1 = (const float*)d_in[4];
  const float* fcb1 = (const float*)d_in[5];
  const float* fcWf = (const float*)d_in[6];
  const float* fcbf = (const float*)d_in[7];
  const float* gW0  = (const float*)d_in[8];
  const float* gb0  = (const float*)d_in[9];
  const float* gW1  = (const float*)d_in[10];
  const float* gb1  = (const float*)d_in[11];
  const float* gW2  = (const float*)d_in[12];
  const float* gb2  = (const float*)d_in[13];
  const float* gWf  = (const float*)d_in[14];
  const float* gbf  = (const float*)d_in[15];

  // workspace layout: emb_h (bf16, N x 128) | perlog (f32, N x 8)
  unsigned short* embh = (unsigned short*)d_ws;
  float* perlog = (float*)((char*)d_ws + (size_t)N_ * H_ * 2);

  float* resid  = (float*)d_out;                 // (B, TL, D) = 261120 f32
  float* outlog = resid + (size_t)N_ * D_;       // (B,)       = 64 f32

  constexpr size_t FC_LDS =
      (size_t)(128 * 96 + 128 * 128 + 128 * 128 + 128 * 128) * 2 + 128 * 4;
  constexpr size_t G_LDS =
      (size_t)(128 * 160 + 256 * 128 + 256 * 128 + 128 * 160) * 2 + 2 * 128 * 4;

  fc_mlp_kernel<<<TILES_, 256, FC_LDS, stream>>>(
      emb, fcW0, fcb0, fcW1, fcb1, fcWf, fcbf, embh);
  g_mlp_jvp_kernel<<<dim3(TILES_, D_), 256, G_LDS, stream>>>(
      x, embh, gW0, gb0, gW1, gb1, gW2, gb2, gWf, gbf, resid, perlog);
  logdet_reduce_kernel<<<B_, 256, 0, stream>>>(perlog, outlog);
}
